// GatedNetwork_31061203484850
// MI455X (gfx1250) — compile-verified, matrix operand order checked by reading, not run on hardware
//
#include <hip/hip_runtime.h>
#include <hip/hip_bf16.h>

// ---------------------------------------------------------------------------
// GatedNetwork for MI455X (gfx1250, wave32, WMMA bf16 16x16x32, f32 accum)
// Double-buffered LDS mainloop; async global->LDS (ASYNCcnt) when available.
// ---------------------------------------------------------------------------

#define NODES   131072
#define CH      256
#define NC      ((size_t)NODES * CH)   // 33,554,432

typedef __attribute__((ext_vector_type(16))) __bf16 v16bf;
typedef __attribute__((ext_vector_type(8)))  __bf16 v8bf;
typedef __attribute__((ext_vector_type(8)))  float  v8f;
typedef __attribute__((ext_vector_type(4)))  int    v4i;

// -------- async global->LDS (gfx1250) with compile-safe fallback -----------
#if defined(__has_builtin)
# if __has_builtin(__builtin_amdgcn_global_load_async_to_lds_b128) && \
     __has_builtin(__builtin_amdgcn_s_wait_asynccnt)
#  define USE_ASYNC_LDS 1
# endif
#endif
#ifndef USE_ASYNC_LDS
# define USE_ASYNC_LDS 0
#endif

__device__ __forceinline__ void copy16_g2s(__bf16* lds, const __bf16* g) {
#if USE_ASYNC_LDS
    // global_load_async_to_lds_b128 (ASYNCcnt-tracked), 16B per lane.
    // Param 0 is an AS1 (global) v4i*, param 1 the AS3 (LDS) destination.
    __builtin_amdgcn_global_load_async_to_lds_b128(
        (__attribute__((address_space(1))) v4i*)g,
        (__attribute__((address_space(3))) v4i*)lds, 0, 0);
#else
    *(v8bf*)lds = *(const v8bf*)g;
#endif
}

// Build a 16x32 bf16 WMMA fragment (A or B) from a row-major LDS row.
// Per ISA 7.12.2: lane L holds row (L&15); halves j=0..7 come from
// k = 8*(L>>4)+j, halves j=8..15 from k = 16 + 8*(L>>4) + (j-8).
__device__ __forceinline__ v16bf make_frag(const __bf16* rowp, int halfsel) {
    v8bf lo = *(const v8bf*)(rowp + 8 * halfsel);
    v8bf hi = *(const v8bf*)(rowp + 16 + 8 * halfsel);
    v16bf f;
#pragma unroll
    for (int t = 0; t < 8; ++t) { f[t] = lo[t]; f[8 + t] = hi[t]; }
    return f;
}

// ---------------------------------------------------------------------------
// Unified double-buffered GEMM mainloop.
//   MODE 0: edge GEMM, K=1024 over segments {h[row], h[col], e[col], e[row]}
//   MODE 1: gather-col GEMM, X row = h16[coli]
//   MODE 2: direct GEMM,     X row = h16[rowi]
// Tile: 64 rows x 256 cols per 256-thread block; 8 waves as 4(M) x 2(N);
// 8 WMMA 16x16x32 per wave per K-step.
template <int MODE, int KTOT, int WSTRIDE>
__device__ __forceinline__ void gemm_mainloop(
    const __bf16* __restrict__ h16, const __bf16* __restrict__ e16,
    const __bf16* __restrict__ Wsrc, int rowi, int coli,
    __bf16 (&Xs)[2][64][40], __bf16 (&Ws)[2][256][40], v8f (&acc)[8])
{
    const int tid  = threadIdx.x;
    const int lane = tid & 31;
    const int wave = tid >> 5;
    const int wm   = wave & 3;
    const int wn   = wave >> 2;
    const int el   = tid >> 2;        // 64 X-rows, 4 threads each
    const int chk  = tid & 3;         // 4 x 16B chunks per X row-slice
    constexpr int NSTEP = KTOT / 32;

    auto issue = [&](int buf, int ks) {
        const __bf16* src;
        int idx, kin;
        if constexpr (MODE == 0) {
            const int seg = ks >> 8;
            src = (seg < 2) ? h16 : e16;
            idx = (seg == 0 || seg == 3) ? rowi : coli;
            kin = ks & 255;
        } else if constexpr (MODE == 1) {
            src = h16; idx = coli; kin = ks;
        } else {
            src = h16; idx = rowi; kin = ks;
        }
        copy16_g2s(&Xs[buf][el][chk * 8],
                   src + (size_t)idx * 256 + kin + chk * 8);     // 1 async op
        const __bf16* wp = Wsrc + (size_t)tid * WSTRIDE + ks;
#pragma unroll
        for (int c = 0; c < 4; ++c)
            copy16_g2s(&Ws[buf][tid][c * 8], wp + c * 8);        // 4 async ops
        if (ks + 64 < KTOT)
            __builtin_prefetch(Wsrc + (size_t)tid * WSTRIDE + ks + 64, 0, 1);
    };

    issue(0, 0);
    for (int i = 0; i < NSTEP; ++i) {
        const int cur = i & 1;
        if (i + 1 < NSTEP) {
            issue(cur ^ 1, (i + 1) * 32);
#if USE_ASYNC_LDS
            __builtin_amdgcn_s_wait_asynccnt(5);   // cur tile resident (in-order)
#endif
        } else {
#if USE_ASYNC_LDS
            __builtin_amdgcn_s_wait_asynccnt(0);
#endif
        }
        __syncthreads();                           // publish cur tile to all waves

        const int mrow = wm * 16 + (lane & 15);
        const int hs   = lane >> 4;
        v16bf a = make_frag(&Xs[cur][mrow][0], hs);
        v16bf b[8];
#pragma unroll
        for (int t = 0; t < 8; ++t)
            b[t] = make_frag(&Ws[cur][wn * 128 + t * 16 + (lane & 15)][0], hs);
#pragma unroll
        for (int t = 0; t < 8; ++t)
            acc[t] = __builtin_amdgcn_wmma_f32_16x16x32_bf16(
                false, a, false, b[t], (short)0, acc[t], false, false);
        __syncthreads();                           // reads done before overwrite
    }
}

// ---------------------------------------------------------------------------
// Stage 0a: cast h, e to bf16
__global__ __launch_bounds__(256) void conv_he_kernel(
    const float* __restrict__ h, const float* __restrict__ e,
    __bf16* __restrict__ h16, __bf16* __restrict__ e16)
{
    size_t gid = (size_t)blockIdx.x * 256 + threadIdx.x;
    if (gid < NC) {
        h16[gid] = (__bf16)h[gid];
        e16[gid] = (__bf16)e[gid];
    }
}

// Stage 0b: build Wcat[n][k] (k = seg*256+kk, seg in {A,B,C,D}), Vw16, Uw16,
// summed bias, and zero the reduction buffers (re-zeroed every call).
__global__ __launch_bounds__(256) void conv_w_kernel(
    const float* __restrict__ Aw, const float* __restrict__ Bw,
    const float* __restrict__ Cw, const float* __restrict__ Dw,
    const float* __restrict__ Vwf, const float* __restrict__ Uwf,
    const float* __restrict__ Ab, const float* __restrict__ Bb,
    const float* __restrict__ Cb, const float* __restrict__ Db,
    __bf16* __restrict__ Wcat, __bf16* __restrict__ Vw16, __bf16* __restrict__ Uw16,
    float* __restrict__ bias_sum, float* __restrict__ sumb, float* __restrict__ sqb,
    float* __restrict__ colsum, float* __restrict__ sraw)
{
    int gid = blockIdx.x * 256 + threadIdx.x;
    if (gid < 262144) {                    // Wcat: [256 n][1024 k], gid = n*1024+k
        int n = gid >> 10, k = gid & 1023;
        int seg = k >> 8, kk = k & 255;
        const float* W = (seg == 0) ? Aw : (seg == 1) ? Bw : (seg == 2) ? Cw : Dw;
        Wcat[gid] = (__bf16)W[n * 256 + kk];
    } else if (gid < 327680) {
        int j = gid - 262144;
        Vw16[j] = (__bf16)Vwf[j];
    } else if (gid < 393216) {
        int j = gid - 327680;
        Uw16[j] = (__bf16)Uwf[j];
    }
    if (gid < 256) {
        bias_sum[gid] = Ab[gid] + Bb[gid] + Cb[gid] + Db[gid];
        sumb[gid] = 0.f; sqb[gid] = 0.f; colsum[gid] = 0.f; sraw[gid] = 0.f;
    }
}

// ---------------------------------------------------------------------------
// Stage 1: gathered 4-way edge GEMM + fused bias + bf16 store + BN partials.
__global__ __launch_bounds__(256) void gemm_edge_kernel(
    const __bf16* __restrict__ h16, const __bf16* __restrict__ e16,
    const long long* __restrict__ eidx, const __bf16* __restrict__ Wcat,
    const float* __restrict__ bias_sum, __bf16* __restrict__ eo16,
    float* __restrict__ sumb, float* __restrict__ sqb)
{
    __shared__ __bf16 Xs[2][64][40];
    __shared__ __bf16 Ws[2][256][40];
    __shared__ float  redS[256];
    __shared__ float  redQ[256];

    const int tid   = threadIdx.x;
    const int lane  = tid & 31;
    const int wave  = tid >> 5;
    const int wm    = wave & 3;
    const int wn    = wave >> 2;
    const int ebase = blockIdx.x * 64;

    const int rowi = (int)eidx[ebase + (tid >> 2)];
    const int coli = (int)eidx[NODES + ebase + (tid >> 2)];
    redS[tid] = 0.f;
    redQ[tid] = 0.f;

    v8f acc[8] = {};
    gemm_mainloop<0, 1024, 1024>(h16, e16, Wcat, rowi, coli, Xs, Ws, acc);

    // epilogue: C/D layout lane L: m = 8*(L>>4)+v, n = L&15
#pragma unroll
    for (int t = 0; t < 8; ++t) {
        const int n = wn * 128 + t * 16 + (lane & 15);
        const float bsum = bias_sum[n];
#pragma unroll
        for (int v = 0; v < 8; ++v) {
            const int m = wm * 16 + 8 * (lane >> 4) + v;
            float val = acc[t][v] + bsum;
            eo16[(size_t)(ebase + m) * 256 + n] = (__bf16)val;
            atomicAdd(&redS[n], val);
            atomicAdd(&redQ[n], val * val);
        }
    }
    __syncthreads();
    atomicAdd(&sumb[tid], redS[tid]);
    atomicAdd(&sqb[tid],  redQ[tid]);
}

// Stage 2: BN finalize
__global__ __launch_bounds__(256) void bn_finalize_kernel(
    const float* __restrict__ sumb, const float* __restrict__ sqb,
    const float* __restrict__ gamma, const float* __restrict__ beta,
    float* __restrict__ scale, float* __restrict__ shiftv)
{
    int c = threadIdx.x;
    const float inv = 1.0f / (float)NODES;
    float mu  = sumb[c] * inv;
    float var = sqb[c] * inv - mu * mu;
    float sc  = gamma[c] * rsqrtf(var + 1e-5f);
    scale[c]  = sc;
    shiftv[c] = beta[c] - mu * sc;
}

// Stage 3a: e_new := e
__global__ __launch_bounds__(256) void copy_e_kernel(
    const float* __restrict__ e, float* __restrict__ enew)
{
    size_t gid = (size_t)blockIdx.x * 256 + threadIdx.x;
    if (gid < NC) enew[gid] = e[gid];
}

// Stage 3b: e_new[row[i]] += relu(bn(e_out[i]))
__global__ __launch_bounds__(256) void scatter_kernel(
    const __bf16* __restrict__ eo16, const long long* __restrict__ eidx,
    const float* __restrict__ scale, const float* __restrict__ shiftv,
    float* __restrict__ enew)
{
    const int i = blockIdx.x;
    const int c = threadIdx.x;
    float val = scale[c] * (float)eo16[(size_t)i * 256 + c] + shiftv[c];
    val = fmaxf(val, 0.f);
    const int r = (int)eidx[i];
    atomicAdd(&enew[(size_t)r * 256 + c], val);
}

// Stage 4: in-place sigmoid + per-channel column sums
__global__ __launch_bounds__(256) void sigmoid_colsum_kernel(
    float* __restrict__ enew, float* __restrict__ colsum)
{
    const int c  = threadIdx.x;
    const int r0 = blockIdx.x * 64;
    float local = 0.f;
    for (int r = r0; r < r0 + 64; ++r) {
        size_t o = (size_t)r * 256 + c;
        float s = 1.0f / (1.0f + __expf(-enew[o]));
        enew[o] = s;
        local += s;
    }
    atomicAdd(&colsum[c], local);
}

// ---------------------------------------------------------------------------
// Stage 5: V-GEMM fused with elementwise multiply-reduce into s_raw[256].
__global__ __launch_bounds__(256) void gemm_v_kernel(
    const __bf16* __restrict__ h16, const long long* __restrict__ eidx,
    const __bf16* __restrict__ Vw, const float* __restrict__ Vb,
    const float* __restrict__ sig, float* __restrict__ sraw)
{
    __shared__ __bf16 Xs[2][64][40];
    __shared__ __bf16 Ws[2][256][40];
    __shared__ float  red[256];

    const int tid   = threadIdx.x;
    const int lane  = tid & 31;
    const int wave  = tid >> 5;
    const int wm    = wave & 3;
    const int wn    = wave >> 2;
    const int ebase = blockIdx.x * 64;

    const int coli = (int)eidx[NODES + ebase + (tid >> 2)];
    red[tid] = 0.f;

    v8f acc[8] = {};
    gemm_mainloop<1, 256, 256>(h16, h16, Vw, 0, coli, Xs, Ws, acc);

#pragma unroll
    for (int t = 0; t < 8; ++t) {
        const int n = wn * 128 + t * 16 + (lane & 15);
        const float vb = Vb[n];
#pragma unroll
        for (int v = 0; v < 8; ++v) {
            const int m = wm * 16 + 8 * (lane >> 4) + v;
            float val = acc[t][v] + vb;
            float sg  = sig[(size_t)(ebase + m) * 256 + n];
            atomicAdd(&red[n], val * sg);
        }
    }
    __syncthreads();
    atomicAdd(&sraw[tid], red[tid]);
}

// Stage 6a: shift2 = U_b + s_raw / (colsum + eps)
__global__ __launch_bounds__(256) void shift2_kernel(
    const float* __restrict__ Ub, const float* __restrict__ sraw,
    const float* __restrict__ colsum, float* __restrict__ shift2)
{
    int c = threadIdx.x;
    shift2[c] = Ub[c] + sraw[c] / (colsum[c] + 1e-5f);
}

// Stage 6b: e_new /= (colsum + eps)  (final e_new output)
__global__ __launch_bounds__(256) void divide_kernel(
    float* __restrict__ enew, const float* __restrict__ colsum)
{
    size_t gid = (size_t)blockIdx.x * 256 + threadIdx.x;
    if (gid < NC) {
        int c = (int)(gid & 255);
        enew[gid] = enew[gid] / (colsum[c] + 1e-5f);
    }
}

// Stage 7: h_out = relu(h @ Uw.T + shift2)
__global__ __launch_bounds__(256) void gemm_u_kernel(
    const __bf16* __restrict__ h16, const __bf16* __restrict__ Uw,
    const float* __restrict__ shift2, float* __restrict__ hout)
{
    __shared__ __bf16 Xs[2][64][40];
    __shared__ __bf16 Ws[2][256][40];

    const int tid   = threadIdx.x;
    const int lane  = tid & 31;
    const int wave  = tid >> 5;
    const int wm    = wave & 3;
    const int wn    = wave >> 2;
    const int mbase = blockIdx.x * 64;

    v8f acc[8] = {};
    gemm_mainloop<2, 256, 256>(h16, h16, Uw, mbase + (tid >> 2), 0, Xs, Ws, acc);

#pragma unroll
    for (int t = 0; t < 8; ++t) {
        const int n = wn * 128 + t * 16 + (lane & 15);
        const float sh = shift2[n];
#pragma unroll
        for (int v = 0; v < 8; ++v) {
            const int m = wm * 16 + 8 * (lane >> 4) + v;
            hout[(size_t)(mbase + m) * 256 + n] = fmaxf(acc[t][v] + sh, 0.f);
        }
    }
}

// ---------------------------------------------------------------------------
extern "C" void kernel_launch(void* const* d_in, const int* in_sizes, int n_in,
                              void* d_out, int out_size, void* d_ws, size_t ws_size,
                              hipStream_t stream) {
    const float*     h    = (const float*)d_in[0];
    const long long* eidx = (const long long*)d_in[1];   // int64 [2, N]
    const float*     e    = (const float*)d_in[2];
    const float* Aw = (const float*)d_in[3];  const float* Ab = (const float*)d_in[4];
    const float* Bw = (const float*)d_in[5];  const float* Bb = (const float*)d_in[6];
    const float* Cw = (const float*)d_in[7];  const float* Cb = (const float*)d_in[8];
    const float* Dw = (const float*)d_in[9];  const float* Db = (const float*)d_in[10];
    const float* gamma = (const float*)d_in[11];
    const float* beta  = (const float*)d_in[12];
    const float* Uwf = (const float*)d_in[13]; const float* Ub = (const float*)d_in[14];
    const float* Vwf = (const float*)d_in[15]; const float* Vb = (const float*)d_in[16];

    float* hout = (float*)d_out;         // [N, C]
    float* enew = hout + NC;             // [N, C] (also used as f32 staging)

    // workspace carve-up (~200 MB)
    __bf16* h16  = (__bf16*)d_ws;
    __bf16* e16  = h16 + NC;
    __bf16* eo16 = e16 + NC;
    __bf16* Wcat = eo16 + NC;            // [256][1024]
    __bf16* Vw16 = Wcat + 256 * 1024;    // [256][256]
    __bf16* Uw16 = Vw16 + 65536;
    float*  fbuf = (float*)(Uw16 + 65536);
    float* bias_sum = fbuf;
    float* sumb   = fbuf + 256;
    float* sqb    = fbuf + 512;
    float* scale  = fbuf + 768;
    float* shiftv = fbuf + 1024;
    float* colsum = fbuf + 1280;
    float* sraw   = fbuf + 1536;
    float* shift2 = fbuf + 1792;

    const int ew_blocks = NODES / 64;          // 2048
    const int el_blocks = (int)(NC / 256);     // 131072

    conv_he_kernel<<<el_blocks, 256, 0, stream>>>(h, e, h16, e16);
    conv_w_kernel<<<1536, 256, 0, stream>>>(Aw, Bw, Cw, Dw, Vwf, Uwf,
                                            Ab, Bb, Cb, Db,
                                            Wcat, Vw16, Uw16,
                                            bias_sum, sumb, sqb, colsum, sraw);
    gemm_edge_kernel<<<ew_blocks, 256, 0, stream>>>(h16, e16, eidx, Wcat,
                                                    bias_sum, eo16, sumb, sqb);
    bn_finalize_kernel<<<1, 256, 0, stream>>>(sumb, sqb, gamma, beta, scale, shiftv);
    copy_e_kernel<<<el_blocks, 256, 0, stream>>>(e, enew);
    scatter_kernel<<<NODES, 256, 0, stream>>>(eo16, eidx, scale, shiftv, enew);
    sigmoid_colsum_kernel<<<ew_blocks, 256, 0, stream>>>(enew, colsum);
    gemm_v_kernel<<<ew_blocks, 256, 0, stream>>>(h16, eidx, Vw16, Vb, enew, sraw);
    shift2_kernel<<<1, 256, 0, stream>>>(Ub, sraw, colsum, shift2);
    divide_kernel<<<el_blocks, 256, 0, stream>>>(enew, colsum);
    gemm_u_kernel<<<ew_blocks, 256, 0, stream>>>(h16, Uw16, shift2, hout);
}
